// GraphSAGEModel_48687749267968
// MI455X (gfx1250) — compile-verified, matrix operand order checked by reading, not run on hardware
//
#include <hip/hip_runtime.h>

// ---------------------------------------------------------------------------
// GraphSAGE 2-layer inference for MI455X (gfx1250, wave32, WMMA + TDM).
//   h = relu( mean_agg(x) @ Wl0 + x @ Wr0 + b0 )
//   y = relu( mean_agg(h) @ Wl1 + h @ Wr1 + b1 )
// GEMMs run on v_wmma_f32_16x16x32_f16; A tiles staged to LDS via the
// Tensor Data Mover (tensor_load_to_lds + s_wait_tensorcnt).
// ---------------------------------------------------------------------------

typedef __attribute__((ext_vector_type(16))) _Float16     v16h;
typedef __attribute__((ext_vector_type(8)))  _Float16     v8h;
typedef __attribute__((ext_vector_type(8)))  float        v8f;
typedef __attribute__((ext_vector_type(4)))  unsigned int u32x4;
typedef __attribute__((ext_vector_type(8)))  int          i32x8;
typedef __attribute__((ext_vector_type(4)))  int          i32x4;

#define NNODES 100000
#define NEDGES 625000
#define DIN    128
#define DHID   256

#if __has_builtin(__builtin_amdgcn_tensor_load_to_lds) && \
    __has_builtin(__builtin_amdgcn_s_wait_tensorcnt)
#define USE_TDM 1
#else
#define USE_TDM 0
#endif

// ------------------------------- scatter ops -------------------------------

__global__ void count_deg(const int* __restrict__ dst, float* __restrict__ cnt, int E) {
    int e = blockIdx.x * blockDim.x + threadIdx.x;
    if (e < E) atomicAdd(&cnt[dst[e]], 1.0f);
}

// agg[dst] += x[src]  (f32 source, 4 floats per thread, Dv = D/4 = 1<<shift)
__global__ void scatter_add_f32(const float* __restrict__ x,
                                const int* __restrict__ src,
                                const int* __restrict__ dst,
                                float* __restrict__ agg,
                                int E, int shift, int D) {
    int t = blockIdx.x * blockDim.x + threadIdx.x;
    int Dv = 1 << shift;
    if (t >= E * Dv) return;
    int e = t >> shift;
    int f = (t & (Dv - 1)) * 4;
    int s = src[e], d = dst[e];
    const float* ip = x + (size_t)s * D + f;
    float*       op = agg + (size_t)d * D + f;
    atomicAdd(op + 0, ip[0]);
    atomicAdd(op + 1, ip[1]);
    atomicAdd(op + 2, ip[2]);
    atomicAdd(op + 3, ip[3]);
}

// agg[dst] += h[src]  (f16 source with row stride ldh)
__global__ void scatter_add_f16(const _Float16* __restrict__ h, int ldh,
                                const int* __restrict__ src,
                                const int* __restrict__ dst,
                                float* __restrict__ agg,
                                int E, int shift, int D) {
    int t = blockIdx.x * blockDim.x + threadIdx.x;
    int Dv = 1 << shift;
    if (t >= E * Dv) return;
    int e = t >> shift;
    int f = (t & (Dv - 1)) * 4;
    int s = src[e], d = dst[e];
    const _Float16* ip = h + (size_t)s * ldh + f;
    float*          op = agg + (size_t)d * D + f;
    atomicAdd(op + 0, (float)ip[0]);
    atomicAdd(op + 1, (float)ip[1]);
    atomicAdd(op + 2, (float)ip[2]);
    atomicAdd(op + 3, (float)ip[3]);
}

// out[node, f..f+3] = (f16)( agg[node, f..f+3] / max(cnt[node],1) )
__global__ void finalize_mean(const float* __restrict__ agg,
                              const float* __restrict__ cnt,
                              _Float16* __restrict__ out,
                              int lda, int ldo, int shift, int total) {
    int t = blockIdx.x * blockDim.x + threadIdx.x;
    if (t >= total) return;
    int Dv = 1 << shift;
    int node = t >> shift;
    int f = (t & (Dv - 1)) * 4;
    float inv = 1.0f / fmaxf(cnt[node], 1.0f);
    const float* ip = agg + (size_t)node * lda + f;
    _Float16*    op = out + (size_t)node * ldo + f;
    op[0] = (_Float16)(ip[0] * inv);
    op[1] = (_Float16)(ip[1] * inv);
    op[2] = (_Float16)(ip[2] * inv);
    op[3] = (_Float16)(ip[3] * inv);
}

// out[node, f..f+3] = (f16) in[node, f..f+3]
__global__ void cvt_f32_f16(const float* __restrict__ in,
                            _Float16* __restrict__ out,
                            int ldi, int ldo, int shift, int total) {
    int t = blockIdx.x * blockDim.x + threadIdx.x;
    if (t >= total) return;
    int Dv = 1 << shift;
    int node = t >> shift;
    int f = (t & (Dv - 1)) * 4;
    const float* ip = in + (size_t)node * ldi + f;
    _Float16*    op = out + (size_t)node * ldo + f;
    op[0] = (_Float16)ip[0];
    op[1] = (_Float16)ip[1];
    op[2] = (_Float16)ip[2];
    op[3] = (_Float16)ip[3];
}

// ------------------------------ weight packing -----------------------------
// Pack [Wl; Wr] (KT x 256, KT = 2*K1) into per-lane WMMA B fragments:
// PW[((kt*16 + nt)*32 + lane)*16 + e] = W[k(kt,lane,e)][nt*16 + (lane&15)]
// where kb = (lane>>4)*8;  e<8 -> k = kt*32+kb+e;  e>=8 -> k = kt*32+kb+8+e.
__global__ void pack_weights(const float* __restrict__ Wl,
                             const float* __restrict__ Wr,
                             _Float16* __restrict__ PW,
                             int K1, int KT) {
    int t = blockIdx.x * blockDim.x + threadIdx.x;
    int total = (KT / 32) * 16 * 32;
    if (t >= total) return;
    int lane = t & 31;
    int nt   = (t >> 5) & 15;
    int kt   = t >> 9;
    int col  = nt * 16 + (lane & 15);
    int kb   = (lane >> 4) * 8;
    _Float16* o = PW + (size_t)t * 16;
    for (int e = 0; e < 16; ++e) {
        int k = kt * 32 + kb + (e < 8 ? e : 8 + e);
        const float* W = (k < K1) ? Wl : Wr;
        int kk = (k < K1) ? k : k - K1;
        o[e] = (_Float16)W[(size_t)kk * DHID + col];
    }
}

// --------------------------------- WMMA GEMM -------------------------------
// Block = 256 threads = 8 wave32 waves. Block computes 16 rows x N=256.
// A tile (16 x KTOT f16) staged once into LDS via TDM; each wave owns 2
// N-tiles sharing one A fragment per K-step (2 back-to-back WMMAs).
template <int KTOT, bool F16OUT>
__global__ __launch_bounds__(256)
void sage_gemm(const _Float16* __restrict__ Acat,   // NNODES x KTOT, f16
               const _Float16* __restrict__ PW,     // packed B fragments
               const float* __restrict__ bias,      // 256
               float* __restrict__ outf,            // f32 out (NNODES x 256)
               _Float16* __restrict__ outh) {       // f16 out (+256, stride 512)
    __shared__ __align__(16) _Float16 As[16 * KTOT];

    const size_t base = (size_t)blockIdx.x * 16 * KTOT;

#if USE_TDM
    if (threadIdx.x < 32) {
        // Tensor DMA descriptor (ISA 08_async_tensor §8): 2D tile,
        // 16 rows x KTOT f16 elements, row stride KTOT, into LDS offset 0
        // (As is the only LDS allocation in this kernel).
        const unsigned long long ga = (unsigned long long)(Acat + base);
        u32x4 g0;
        g0[0] = 1u;                                   // count=1, user mode
        g0[1] = 0u;                                   // lds_addr = 0
        g0[2] = (unsigned)(ga & 0xFFFFFFFFu);         // global_addr[31:0]
        g0[3] = (unsigned)((ga >> 32) & 0x1FFFFFFu)   // global_addr[56:32]
              | (2u << 30);                           // type = 2 ("image")
        const unsigned td0 = KTOT, td1 = 16, tile0 = KTOT, tile1 = 16;
        i32x8 g1;
        g1[0] = (int)(1u << 16);                      // data_size = 1 (2 bytes)
        g1[1] = (int)((td0 & 0xFFFFu) << 16);         // tensor_dim0[15:0]
        g1[2] = (int)((td0 >> 16) | ((td1 & 0xFFFFu) << 16));
        g1[3] = (int)((td1 >> 16) | (tile0 << 16));   // tile_dim0
        g1[4] = (int)tile1;                           // tile_dim1, tile_dim2=0
        g1[5] = (int)td0;                             // tensor_dim0_stride lo
        g1[6] = 0;                                    // stride hi, dim1_stride
        g1[7] = 0;
        i32x4 g2 = {0, 0, 0, 0};
        i32x4 g3 = {0, 0, 0, 0};
#if defined(__clang_major__) && (__clang_major__ >= 23)
        i32x8 g4 = {0, 0, 0, 0, 0, 0, 0, 0};
        __builtin_amdgcn_tensor_load_to_lds(g0, g1, g2, g3, g4, 0);
#else
        __builtin_amdgcn_tensor_load_to_lds(g0, g1, g2, g3, 0);
#endif
        __builtin_amdgcn_s_wait_tensorcnt(0);
    }
#else
    for (int i = threadIdx.x; i < 16 * KTOT / 8; i += 256) {
        *(v8h*)(As + i * 8) = *(const v8h*)(Acat + base + i * 8);
    }
#endif
    __syncthreads();

    const int lane = threadIdx.x & 31;
    const int wid  = threadIdx.x >> 5;
    const int row  = lane & 15;              // A fragment row (M within tile)
    const int kb   = (lane >> 4) * 8;        // K sub-offset per half-wave
    const int gm   = blockIdx.x * 16;
    const int nt0  = wid * 2;

    __builtin_prefetch(PW + ((size_t)nt0 * 32 + lane) * 16, 0, 3);

    v8f c0 = {};
    v8f c1 = {};
    for (int kt = 0; kt < KTOT / 32; ++kt) {
        union { v16h v; v8h h[2]; } a, b0, b1;
        const _Float16* ap = As + row * KTOT + kt * 32 + kb;
        a.h[0] = *(const v8h*)(ap);          // K = kb..kb+7
        a.h[1] = *(const v8h*)(ap + 16);     // K = kb+16..kb+23
        const _Float16* bp = PW + ((size_t)(kt * 16 + nt0) * 32 + lane) * 16;
        b0.h[0] = *(const v8h*)(bp);
        b0.h[1] = *(const v8h*)(bp + 8);
        b1.h[0] = *(const v8h*)(bp + 512);   // next N-tile: +32 lanes * 16 halves
        b1.h[1] = *(const v8h*)(bp + 520);
        c0 = __builtin_amdgcn_wmma_f32_16x16x32_f16(
                 false, a.v, false, b0.v, (short)0, c0, false, false);
        c1 = __builtin_amdgcn_wmma_f32_16x16x32_f16(
                 false, a.v, false, b1.v, (short)0, c1, false, false);
    }

    // C/D layout: VGPR r -> M = (lane>>4)*8 + r, N = lane&15 (+ nt*16)
    const int rbase = (lane >> 4) * 8;
    const int col0  = nt0 * 16 + (lane & 15);
    for (int half = 0; half < 2; ++half) {
        const v8f& c = half ? c1 : c0;
        const int col = col0 + half * 16;
        const float bv = bias[col];
        if (F16OUT) {
            _Float16* p = outh + (size_t)(gm + rbase) * 512 + 256 + col;
            for (int r = 0; r < 8; ++r) {
                float v = c[r] + bv;
                p[(size_t)r * 512] = (_Float16)(v > 0.0f ? v : 0.0f);
            }
        } else {
            float* p = outf + (size_t)(gm + rbase) * 256 + col;
            for (int r = 0; r < 8; ++r) {
                float v = c[r] + bv;
                p[(size_t)r * 256] = (v > 0.0f ? v : 0.0f);
            }
        }
    }
}

// --------------------------------- launcher --------------------------------

extern "C" void kernel_launch(void* const* d_in, const int* in_sizes, int n_in,
                              void* d_out, int out_size, void* d_ws, size_t ws_size,
                              hipStream_t stream) {
    const float* x   = (const float*)d_in[0];
    const int*   ei  = (const int*)d_in[1];     // [2, E] int32
    const float* Wl0 = (const float*)d_in[2];
    const float* Wr0 = (const float*)d_in[3];
    const float* b0  = (const float*)d_in[4];
    const float* Wl1 = (const float*)d_in[5];
    const float* Wr1 = (const float*)d_in[6];
    const float* b1  = (const float*)d_in[7];
    const int* src = ei;
    const int* dst = ei + NEDGES;

    // workspace layout
    char* ws = (char*)d_ws;
    size_t off = 0;
    float* agg = (float*)(ws + off); off += (size_t)NNODES * DHID * 4;   // 102.4 MB
    float* cnt = (float*)(ws + off); off += (size_t)NNODES * 4;          // 0.4 MB
    off = (off + 255) & ~(size_t)255;
    _Float16* Acat0 = (_Float16*)(ws + off); off += (size_t)NNODES * 256 * 2;  // [agg0 | x]
    _Float16* Acat1 = (_Float16*)(ws + off); off += (size_t)NNODES * 512 * 2;  // [agg1 | h]
    _Float16* PW0   = (_Float16*)(ws + off); off += (size_t)(256 / 32) * 16 * 32 * 16 * 2;
    _Float16* PW1   = (_Float16*)(ws + off); off += (size_t)(512 / 32) * 16 * 32 * 16 * 2;

    // zero accumulators (capture-safe)
    hipMemsetAsync(agg, 0, (size_t)NNODES * DIN * 4, stream);
    hipMemsetAsync(cnt, 0, (size_t)NNODES * 4, stream);

    // degrees + weight packing + x -> f16
    count_deg<<<(NEDGES + 255) / 256, 256, 0, stream>>>(dst, cnt, NEDGES);
    pack_weights<<<(4096 + 255) / 256, 256, 0, stream>>>(Wl0, Wr0, PW0, 128, 256);
    pack_weights<<<(8192 + 255) / 256, 256, 0, stream>>>(Wl1, Wr1, PW1, 256, 512);
    {
        int total = NNODES * (DIN / 4);  // 3.2M
        cvt_f32_f16<<<(total + 255) / 256, 256, 0, stream>>>(x, Acat0 + DIN, DIN, 256, 5, total);
    }

    // ---- layer 0 ----
    {
        int total = NEDGES * (DIN / 4);  // 20M
        scatter_add_f32<<<(total + 255) / 256, 256, 0, stream>>>(x, src, dst, agg, NEDGES, 5, DIN);
    }
    {
        int total = NNODES * (DIN / 4);
        finalize_mean<<<(total + 255) / 256, 256, 0, stream>>>(agg, cnt, Acat0, DIN, 256, 5, total);
    }
    // h (relu, f16) -> Acat1[:, 256:512]
    sage_gemm<256, true><<<NNODES / 16, 256, 0, stream>>>(Acat0, PW0, b0, nullptr, Acat1);

    // ---- layer 1 ----
    hipMemsetAsync(agg, 0, (size_t)NNODES * DHID * 4, stream);
    {
        int total = NEDGES * (DHID / 4);  // 40M
        scatter_add_f16<<<(total + 255) / 256, 256, 0, stream>>>(Acat1 + 256, 512, src, dst, agg,
                                                                 NEDGES, 6, DHID);
    }
    {
        int total = NNODES * (DHID / 4);
        finalize_mean<<<(total + 255) / 256, 256, 0, stream>>>(agg, cnt, Acat1, DHID, 512, 6, total);
    }
    sage_gemm<512, false><<<NNODES / 16, 256, 0, stream>>>(Acat1, PW1, b1, (float*)d_out, nullptr);
}
